// Block_8177617731884
// MI455X (gfx1250) — matrix operand order repaired and checked by reading, not compiled, not run on hardware
//
#include <hip/hip_runtime.h>

// ---------------------------------------------------------------------------
// MI455X (gfx1250, wave32) transformer block: rmsnorm -> attn -> rmsnorm -> MoE
// GEMMs: v_wmma_f32_16x16x32_f16 with double-buffered async LDS staging
// (global_load_async_to_lds_b128 + s_wait_asynccnt pipeline).
// ---------------------------------------------------------------------------

typedef __attribute__((ext_vector_type(16))) _Float16 v16h;
typedef __attribute__((ext_vector_type(8)))  float    v8f;

#define N_EMBD     1024
#define N_HEAD     16
#define HEAD_DIM   64
#define N_EXPERTS  8
#define EXPERT_DIM 2048
#define BATCH      2
#define SEQ        2048
#define NTOK       (BATCH * SEQ)   // 4096

union FragU { v16h v; unsigned u[8]; };

__device__ __forceinline__ v8f wmma_f16(v16h a, v16h b, v8f c) {
  // D = A(16x32 f16) * B(32x16 f16) + C(16x16 f32)
  return __builtin_amdgcn_wmma_f32_16x16x32_f16(false, a, false, b, (short)0, c,
                                                false, false);
}

// Async copy of 16 bytes: global -> LDS, tracked by ASYNCcnt (per-lane).
// LDS address = low 32 bits of generic pointer (== LDS offset per aperture map).
__device__ __forceinline__ void async_copy16(const _Float16* g, _Float16* l) {
  unsigned lds_addr = (unsigned)(size_t)l;
  unsigned long long gaddr = (unsigned long long)(size_t)g;
  asm volatile("global_load_async_to_lds_b128 %0, %1, off"
               :: "v"(lds_addr), "v"(gaddr) : "memory");
}
__device__ __forceinline__ void wait_asynccnt_0() {
  asm volatile("s_wait_asynccnt 0x0" ::: "memory");
}
__device__ __forceinline__ void wait_asynccnt_4() {
  asm volatile("s_wait_asynccnt 0x4" ::: "memory");
}

// A fragment (16x32, f16) from row-major [M][K] source, leading dim ld (halfs).
// ISA layout: lanes 0-15 -> M=lane, K pairs {0,1..6,7}+{16..23}; lanes 16-31 +8 K offset.
__device__ __forceinline__ v16h load_frag_a(const _Float16* base, int ld) {
  const int lane = threadIdx.x & 31;
  const int hs = lane >> 4;
  const int m  = lane & 15;
  FragU f;
#pragma unroll
  for (int v = 0; v < 8; ++v) {
    int k = (v < 4) ? (2 * v + 8 * hs) : (16 + 2 * (v - 4) + 8 * hs);
    f.u[v] = *reinterpret_cast<const unsigned*>(base + m * ld + k);
  }
  return f.v;
}

// B fragment (K=32 x N=16, f16) from B^T row-major [N][K] source, leading dim ld.
// ISA layout: lanes 0-15 hold K=0-7,16-23; lanes 16-31 +8 K offset; N = lane & 15.
__device__ __forceinline__ v16h load_frag_b(const _Float16* base, int ld) {
  const int lane = threadIdx.x & 31;
  const int n  = lane & 15;
  const int kb = (lane >> 4) * 8;
  FragU f;
#pragma unroll
  for (int v = 0; v < 8; ++v)
    f.u[v] = *reinterpret_cast<const unsigned*>(base + n * ld + kb + ((v < 4) ? 2 * v : 16 + 2 * (v - 4)));
  return f.v;
}

// ---------------------------------------------------------------------------
// RMSNorm (f32 in) -> f16 out
// ---------------------------------------------------------------------------
__global__ __launch_bounds__(256) void rmsnorm_f16_kernel(
    const float* __restrict__ in, _Float16* __restrict__ out) {
  __shared__ float red[256];
  const int row = blockIdx.x;
  const float* xr = in + (size_t)row * N_EMBD;
  float ss = 0.f;
#pragma unroll
  for (int i = threadIdx.x; i < N_EMBD; i += 256) { float v = xr[i]; ss += v * v; }
  red[threadIdx.x] = ss;
  __syncthreads();
  for (int s = 128; s > 0; s >>= 1) {
    if ((int)threadIdx.x < s) red[threadIdx.x] += red[threadIdx.x + s];
    __syncthreads();
  }
  const float inv = rsqrtf(red[0] * (1.0f / N_EMBD) + 1e-5f);
#pragma unroll
  for (int i = threadIdx.x; i < N_EMBD; i += 256)
    out[(size_t)row * N_EMBD + i] = (_Float16)(xr[i] * inv);
}

// ---------------------------------------------------------------------------
// Transpose-convert: f32 [K][N] -> f16 [N][K]  (B^T staging for WMMA)
// ---------------------------------------------------------------------------
__global__ void transpose_f32_to_f16_kernel(const float* __restrict__ in,
                                            _Float16* __restrict__ out,
                                            int K, int N) {
  int idx = blockIdx.x * 256 + threadIdx.x;
  int total = K * N;
  if (idx >= total) return;
  int n = idx / K;
  int k = idx - n * K;
  out[idx] = (_Float16)in[(size_t)k * N + n];
}

// ---------------------------------------------------------------------------
// Tiled WMMA GEMM: C[M][N] = A[M][K] * B[K][N], A f16 row-major, B as B^T f16.
// Block 256 thr = 8 waves, tile 128x128, wave tile 32x64 (2x4 WMMA frags).
// Double-buffered LDS, filled by async-to-LDS DMA (ASYNCcnt pipeline):
//   issue(t+1) -> s_wait_asynccnt 4 (in-order done => tile t landed) -> barrier
//   -> 8 WMMAs from LDS -> barrier.
// ---------------------------------------------------------------------------
enum { GEMM_F16OUT = 0, GEMM_F16OUT_SQRELU = 1, GEMM_RESID_F32 = 2, GEMM_SCALED_ACC = 3 };

template <int MODE>
__global__ __launch_bounds__(256) void gemm_wmma_kernel(
    const _Float16* __restrict__ A, const _Float16* __restrict__ BT,
    _Float16* __restrict__ C16, float* __restrict__ C32,
    const float* __restrict__ resid,
    const float* __restrict__ rowscale, int rs_stride,
    int M, int N, int K) {
  __shared__ _Float16 sA[2][128][48];   // 12 KB per buffer (96B rows, 16B aligned)
  __shared__ _Float16 sB[2][128][48];
  const int tid  = threadIdx.x;
  const int lane = tid & 31, wave = tid >> 5;
  const int wm = (wave & 3) * 32;    // 0..96
  const int wn = (wave >> 2) * 64;   // 0,64
  const int bm = blockIdx.x * 128;
  const int bn = blockIdx.y * 128;

  // Each thread owns 2 A-chunks + 2 B-chunks (16B each) per K-tile: 4 async ops.
  auto issue_tile = [&](int buf, int k0) {
#pragma unroll
    for (int c = tid; c < 512; c += 256) {   // 128 rows x 4 chunks
      int r = c >> 2, ch = c & 3;
      async_copy16(A + (size_t)(bm + r) * K + k0 + ch * 8, &sA[buf][r][ch * 8]);
      async_copy16(BT + (size_t)(bn + r) * K + k0 + ch * 8, &sB[buf][r][ch * 8]);
    }
  };

  v8f acc[2][4] = {};
  const int nk = K >> 5;
  issue_tile(0, 0);
  for (int t = 0; t < nk; ++t) {
    const int cur = t & 1;
    if (t + 1 < nk) {
      issue_tile(cur ^ 1, (t + 1) << 5);
      wait_asynccnt_4();   // 4 newest (next tile) may remain in flight
    } else {
      wait_asynccnt_0();
    }
    __syncthreads();       // all threads' tile-t transfers have landed

    v16h a0 = load_frag_a(&sA[cur][wm][0], 48);
    v16h a1 = load_frag_a(&sA[cur][wm + 16][0], 48);
    v16h b0 = load_frag_b(&sB[cur][wn][0], 48);
    v16h b1 = load_frag_b(&sB[cur][wn + 16][0], 48);
    v16h b2 = load_frag_b(&sB[cur][wn + 32][0], 48);
    v16h b3 = load_frag_b(&sB[cur][wn + 48][0], 48);
    acc[0][0] = wmma_f16(a0, b0, acc[0][0]);
    acc[0][1] = wmma_f16(a0, b1, acc[0][1]);
    acc[0][2] = wmma_f16(a0, b2, acc[0][2]);
    acc[0][3] = wmma_f16(a0, b3, acc[0][3]);
    acc[1][0] = wmma_f16(a1, b0, acc[1][0]);
    acc[1][1] = wmma_f16(a1, b1, acc[1][1]);
    acc[1][2] = wmma_f16(a1, b2, acc[1][2]);
    acc[1][3] = wmma_f16(a1, b3, acc[1][3]);
    __syncthreads();       // done reading buf cur before it is refilled
  }

  const int chn = lane & 15, rh = (lane >> 4) * 8;   // C/D layout
#pragma unroll
  for (int mi = 0; mi < 2; ++mi)
#pragma unroll
    for (int ni = 0; ni < 4; ++ni)
#pragma unroll
      for (int i = 0; i < 8; ++i) {
        size_t row = (size_t)(bm + wm + mi * 16 + rh + i);
        size_t col = (size_t)(bn + wn + ni * 16 + chn);
        size_t idx = row * (size_t)N + col;
        float val = acc[mi][ni][i];
        if constexpr (MODE == GEMM_F16OUT) {
          C16[idx] = (_Float16)val;
        } else if constexpr (MODE == GEMM_F16OUT_SQRELU) {
          float r = fmaxf(val, 0.f);
          C16[idx] = (_Float16)(r * r);
        } else if constexpr (MODE == GEMM_RESID_F32) {
          C32[idx] = val + resid[idx];
        } else {  // GEMM_SCALED_ACC
          C32[idx] += val * rowscale[row * rs_stride];
        }
      }
}

// ---------------------------------------------------------------------------
// Flash attention with WMMA. Block = 4 waves (128 thr), 64 queries/block,
// 16 queries/wave, key tiles of 32, online softmax, causal.
// ---------------------------------------------------------------------------
__global__ __launch_bounds__(128) void attention_kernel(
    const _Float16* __restrict__ q16, const _Float16* __restrict__ k16,
    const _Float16* __restrict__ v16, _Float16* __restrict__ o16) {
  __shared__ _Float16 sK[32][80];       // [key][d]  (B^T layout for QK^T)
  __shared__ _Float16 sVt[64][40];      // [d][key]  (B^T layout for P*V)
  __shared__ _Float16 sP[4][16][40];    // per-wave P scratch (C->A layout hop)
  const int tid = threadIdx.x;
  const int lane = tid & 31, wave = tid >> 5;
  const int qblk = blockIdx.x, h = blockIdx.y, b = blockIdx.z;
  const int qbase = qblk * 64 + wave * 16;
  const size_t rowbase = (size_t)b * SEQ * N_EMBD;
  const int chn = lane & 15, rh = (lane >> 4) * 8;
  const float sc = 0.125f;  // 1/sqrt(64)

  const _Float16* qptr = q16 + rowbase + (size_t)qbase * N_EMBD + h * HEAD_DIM;
  v16h aq0 = load_frag_a(qptr, N_EMBD);
  v16h aq1 = load_frag_a(qptr + 32, N_EMBD);

  float m_run[8], l_run[8];
  v8f o[4] = {};
#pragma unroll
  for (int i = 0; i < 8; ++i) { m_run[i] = -1e30f; l_run[i] = 0.f; }

  const int ntiles = qblk * 2 + 2;  // causal: keys up to end of this q-block
  for (int kt = 0; kt < ntiles; ++kt) {
    const int kb = kt * 32;
    for (int c = tid; c < 256; c += 128) {  // 32 rows x 8 chunks of 8 halfs
      int r = c >> 3, ch = c & 7;
      *reinterpret_cast<uint4*>(&sK[r][ch * 8]) = *reinterpret_cast<const uint4*>(
          k16 + rowbase + (size_t)(kb + r) * N_EMBD + h * HEAD_DIM + ch * 8);
      union { uint4 u; _Float16 hh[8]; } vv;
      vv.u = *reinterpret_cast<const uint4*>(
          v16 + rowbase + (size_t)(kb + r) * N_EMBD + h * HEAD_DIM + ch * 8);
#pragma unroll
      for (int t = 0; t < 8; ++t) sVt[ch * 8 + t][r] = vv.hh[t];
    }
    __syncthreads();

    // S[16q x 32k] = Q[16x64] * K^T, split into two 16-key fragments
    v8f s0 = {}, s1 = {};
    s0 = wmma_f16(aq0, load_frag_b(&sK[0][0], 80), s0);
    s0 = wmma_f16(aq1, load_frag_b(&sK[0][32], 80), s0);
    s1 = wmma_f16(aq0, load_frag_b(&sK[16][0], 80), s1);
    s1 = wmma_f16(aq1, load_frag_b(&sK[16][32], 80), s1);

    // scale + causal mask + tile row-max (reduce across 16-lane row group)
    float tmax[8];
#pragma unroll
    for (int i = 0; i < 8; ++i) {
      int qpos = qbase + rh + i;
      float v0 = s0[i] * sc; if (kb + chn > qpos)      v0 = -1e30f;
      float v1 = s1[i] * sc; if (kb + 16 + chn > qpos) v1 = -1e30f;
      s0[i] = v0; s1[i] = v1;
      tmax[i] = fmaxf(v0, v1);
    }
#pragma unroll
    for (int off = 8; off >= 1; off >>= 1)
#pragma unroll
      for (int i = 0; i < 8; ++i)
        tmax[i] = fmaxf(tmax[i], __shfl_xor(tmax[i], off, 32));

    float rsum[8], corr[8];
#pragma unroll
    for (int i = 0; i < 8; ++i) {
      float mn = fmaxf(m_run[i], tmax[i]);
      corr[i] = __expf(m_run[i] - mn);
      float p0 = __expf(s0[i] - mn);
      float p1 = __expf(s1[i] - mn);
      s0[i] = p0; s1[i] = p1;
      rsum[i] = p0 + p1;
      m_run[i] = mn;
    }
#pragma unroll
    for (int off = 8; off >= 1; off >>= 1)
#pragma unroll
      for (int i = 0; i < 8; ++i) rsum[i] += __shfl_xor(rsum[i], off, 32);
#pragma unroll
    for (int i = 0; i < 8; ++i) {
      l_run[i] = l_run[i] * corr[i] + rsum[i];
#pragma unroll
      for (int j = 0; j < 4; ++j) o[j][i] *= corr[i];
    }

    // P: C-layout -> LDS -> A-layout, then O += P * V
#pragma unroll
    for (int i = 0; i < 8; ++i) {
      sP[wave][rh + i][chn]      = (_Float16)s0[i];
      sP[wave][rh + i][16 + chn] = (_Float16)s1[i];
    }
    v16h ap = load_frag_a(&sP[wave][0][0], 40);
    o[0] = wmma_f16(ap, load_frag_b(&sVt[0][0], 40), o[0]);
    o[1] = wmma_f16(ap, load_frag_b(&sVt[16][0], 40), o[1]);
    o[2] = wmma_f16(ap, load_frag_b(&sVt[32][0], 40), o[2]);
    o[3] = wmma_f16(ap, load_frag_b(&sVt[48][0], 40), o[3]);
    __syncthreads();
  }

  _Float16* optr = o16 + rowbase + (size_t)qbase * N_EMBD + h * HEAD_DIM;
#pragma unroll
  for (int j = 0; j < 4; ++j)
#pragma unroll
    for (int i = 0; i < 8; ++i)
      optr[(size_t)(rh + i) * N_EMBD + j * 16 + chn] = (_Float16)(o[j][i] / l_run[i]);
}

// ---------------------------------------------------------------------------
// Router: logits, softmax, top-2, normalized combine weights [NTOK][8].
// One wave per token (tiny: 67 MFLOP).
// ---------------------------------------------------------------------------
__global__ __launch_bounds__(256) void router_kernel(
    const _Float16* __restrict__ h2, const float* __restrict__ rw,
    float* __restrict__ combine) {
  const int wave = threadIdx.x >> 5;
  const int lane = threadIdx.x & 31;
  const int tok = blockIdx.x * 8 + wave;
  const _Float16* hr = h2 + (size_t)tok * N_EMBD;
  float lg[8] = {0.f, 0.f, 0.f, 0.f, 0.f, 0.f, 0.f, 0.f};
  for (int d = lane; d < N_EMBD; d += 32) {
    float hv = (float)hr[d];
#pragma unroll
    for (int e = 0; e < 8; ++e) lg[e] += hv * rw[d * 8 + e];
  }
#pragma unroll
  for (int off = 16; off >= 1; off >>= 1)
#pragma unroll
    for (int e = 0; e < 8; ++e) lg[e] += __shfl_xor(lg[e], off, 32);
  if (lane == 0) {
    int i1 = 0;
#pragma unroll
    for (int e = 1; e < 8; ++e) if (lg[e] > lg[i1]) i1 = e;
    int i2 = (i1 == 0) ? 1 : 0;
#pragma unroll
    for (int e = 0; e < 8; ++e) if (e != i1 && lg[e] > lg[i2]) i2 = e;
    // normalized among the two (softmax denominator cancels)
    float p2 = __expf(lg[i2] - lg[i1]);
    float inv = 1.0f / (1.0f + p2);
#pragma unroll
    for (int e = 0; e < 8; ++e) combine[(size_t)tok * 8 + e] = 0.f;
    combine[(size_t)tok * 8 + i1] = inv;
    combine[(size_t)tok * 8 + i2] = p2 * inv;
  }
}

// ---------------------------------------------------------------------------
// Host orchestration
// ---------------------------------------------------------------------------
extern "C" void kernel_launch(void* const* d_in, const int* in_sizes, int n_in,
                              void* d_out, int out_size, void* d_ws, size_t ws_size,
                              hipStream_t stream) {
  const float* x   = (const float*)d_in[0];
  const float* wq  = (const float*)d_in[1];
  const float* wk  = (const float*)d_in[2];
  const float* wv  = (const float*)d_in[3];
  const float* wo  = (const float*)d_in[4];
  const float* rw  = (const float*)d_in[5];
  const float* fc1 = (const float*)d_in[6];
  const float* fc2 = (const float*)d_in[7];
  float* out = (float*)d_out;

  char* cur = (char*)d_ws;
  auto alloc = [&](size_t bytes) -> char* {
    char* p = cur;
    cur += (bytes + 255) & ~(size_t)255;
    return p;
  };
  _Float16* h16    = (_Float16*)alloc((size_t)NTOK * N_EMBD * 2);       // 8 MB
  _Float16* wT     = (_Float16*)alloc((size_t)N_EMBD * N_EMBD * 2);     // 2 MB (reused)
  _Float16* q16    = (_Float16*)alloc((size_t)NTOK * N_EMBD * 2);       // 8 MB
  _Float16* k16    = (_Float16*)alloc((size_t)NTOK * N_EMBD * 2);       // 8 MB
  _Float16* v16    = (_Float16*)alloc((size_t)NTOK * N_EMBD * 2);       // 8 MB
  _Float16* attn16 = (_Float16*)alloc((size_t)NTOK * N_EMBD * 2);       // 8 MB
  float*    x2     = (float*)   alloc((size_t)NTOK * N_EMBD * 4);       // 16 MB
  _Float16* h2     = (_Float16*)alloc((size_t)NTOK * N_EMBD * 2);       // 8 MB
  float*    comb   = (float*)   alloc((size_t)NTOK * N_EXPERTS * 4);    // 128 KB
  _Float16* fcT    = (_Float16*)alloc((size_t)N_EMBD * EXPERT_DIM * 2); // 4 MB (reused)
  _Float16* H1     = (_Float16*)alloc((size_t)NTOK * EXPERT_DIM * 2);   // 16 MB

  const dim3 gProj(NTOK / 128, N_EMBD / 128);
  const dim3 gFc1(NTOK / 128, EXPERT_DIM / 128);
  const int gTwc = (N_EMBD * N_EMBD) / 256;
  const int gTwe = (N_EMBD * EXPERT_DIM) / 256;

  // --- attention branch ---
  rmsnorm_f16_kernel<<<NTOK, 256, 0, stream>>>(x, h16);

  transpose_f32_to_f16_kernel<<<gTwc, 256, 0, stream>>>(wq, wT, N_EMBD, N_EMBD);
  gemm_wmma_kernel<GEMM_F16OUT><<<gProj, 256, 0, stream>>>(
      h16, wT, q16, nullptr, nullptr, nullptr, 0, NTOK, N_EMBD, N_EMBD);
  transpose_f32_to_f16_kernel<<<gTwc, 256, 0, stream>>>(wk, wT, N_EMBD, N_EMBD);
  gemm_wmma_kernel<GEMM_F16OUT><<<gProj, 256, 0, stream>>>(
      h16, wT, k16, nullptr, nullptr, nullptr, 0, NTOK, N_EMBD, N_EMBD);
  transpose_f32_to_f16_kernel<<<gTwc, 256, 0, stream>>>(wv, wT, N_EMBD, N_EMBD);
  gemm_wmma_kernel<GEMM_F16OUT><<<gProj, 256, 0, stream>>>(
      h16, wT, v16, nullptr, nullptr, nullptr, 0, NTOK, N_EMBD, N_EMBD);

  attention_kernel<<<dim3(SEQ / 64, N_HEAD, BATCH), 128, 0, stream>>>(
      q16, k16, v16, attn16);

  transpose_f32_to_f16_kernel<<<gTwc, 256, 0, stream>>>(wo, wT, N_EMBD, N_EMBD);
  gemm_wmma_kernel<GEMM_RESID_F32><<<gProj, 256, 0, stream>>>(
      attn16, wT, nullptr, x2, x, nullptr, 0, NTOK, N_EMBD, N_EMBD);

  // --- MoE branch ---
  rmsnorm_f16_kernel<<<NTOK, 256, 0, stream>>>(x2, h2);
  router_kernel<<<NTOK / 8, 256, 0, stream>>>(h2, rw, comb);

  // out starts as x2 (residual); experts accumulate weighted contributions.
  hipMemcpyAsync(out, x2, (size_t)NTOK * N_EMBD * sizeof(float),
                 hipMemcpyDeviceToDevice, stream);

  for (int e = 0; e < N_EXPERTS; ++e) {
    transpose_f32_to_f16_kernel<<<gTwe, 256, 0, stream>>>(
        fc1 + (size_t)e * N_EMBD * EXPERT_DIM, fcT, N_EMBD, EXPERT_DIM);
    gemm_wmma_kernel<GEMM_F16OUT_SQRELU><<<gFc1, 256, 0, stream>>>(
        h2, fcT, H1, nullptr, nullptr, nullptr, 0, NTOK, EXPERT_DIM, N_EMBD);
    transpose_f32_to_f16_kernel<<<gTwe, 256, 0, stream>>>(
        fc2 + (size_t)e * EXPERT_DIM * N_EMBD, fcT, EXPERT_DIM, N_EMBD);
    gemm_wmma_kernel<GEMM_SCALED_ACC><<<gProj, 256, 0, stream>>>(
        H1, fcT, nullptr, out, nullptr, comb + e, N_EXPERTS,
        NTOK, N_EMBD, EXPERT_DIM);
  }
}
